// AdvancedParticleGNN_55972013802028
// MI455X (gfx1250) — matrix operand order repaired and checked by reading, not compiled, not run on hardware
//
#include <hip/hip_runtime.h>
#include <hip/hip_bf16.h>
#include <stdint.h>

// ---------------------------------------------------------------------------
// AdvancedParticleGNN forward for MI455X (gfx1250, wave32).
//  - EdgeConv MLP + GAT projection: v_wmma_f32_16x16x32_f16, f32 accumulate.
//  - Edge gather: global_load_async_to_lds_b128 (ASYNCcnt) + double buffering.
//  - Softmax / LN / pooling: fp32 atomics + wave32 shuffle reductions.
// ---------------------------------------------------------------------------

#define N_NODESC 50000
#define N_EDGESC 640000
#define HIDC     128
#define HEADSC   8
#define N_GRAPHSC 128
#define N_CLASSESC 5
#define ORDNEG 0x007FFFFFu   // ordered-uint encoding of -inf

typedef __attribute__((ext_vector_type(16))) _Float16 v16h;
typedef __attribute__((ext_vector_type(8)))  _Float16 v8h;
typedef __attribute__((ext_vector_type(8)))  float    v8f;

union V16U { v16h v; v8h h[2]; };

__device__ __forceinline__ v16h ldfrag(const _Float16* p0, const _Float16* p1) {
  V16U u; u.h[0] = *(const v8h*)p0; u.h[1] = *(const v8h*)p1; return u.v;
}

__device__ __forceinline__ unsigned f2ord(float f) {
  unsigned u = __float_as_uint(f);
  return (u & 0x80000000u) ? ~u : (u | 0x80000000u);
}
__device__ __forceinline__ float ord2f(unsigned u) {
  u = (u & 0x80000000u) ? (u & 0x7FFFFFFFu) : ~u;
  return __uint_as_float(u);
}

// async 16B global -> LDS copy (per active lane); tracked by ASYNCcnt.
__device__ __forceinline__ void async_g2l_b128(void* ldsPtr, const void* gPtr) {
  unsigned ldsOff = (unsigned)(size_t)ldsPtr;          // low 32b = LDS offset
  unsigned long long ga = (unsigned long long)(size_t)gPtr;
  asm volatile("global_load_async_to_lds_b128 %0, %1, off"
               :: "v"(ldsOff), "v"(ga) : "memory");
}

// ------------------------------- fill --------------------------------------
__global__ void k_fill(unsigned* __restrict__ p, unsigned v, size_t n) {
  size_t i = (size_t)blockIdx.x * 256 + threadIdx.x;
  if (i < n) p[i] = v;
}

// -------------------- f32 [K][N] -> f16 transposed [N][K] ------------------
__global__ void k_cvtT(const float* __restrict__ W, _Float16* __restrict__ WT,
                       int K, int Nc) {
  int i = blockIdx.x * 256 + threadIdx.x;
  if (i >= K * Nc) return;
  int k = i / Nc, n = i - k * Nc;
  WT[n * K + k] = (_Float16)W[i];
}

// ------------------------------ encoder ------------------------------------
__global__ __launch_bounds__(128)
void k_encoder(const float* __restrict__ x, const float* __restrict__ W,
               const float* __restrict__ bias,
               const float* __restrict__ bnb, const float* __restrict__ bng,
               const float* __restrict__ bnm, const float* __restrict__ bnv,
               float* __restrict__ h, _Float16* __restrict__ hh) {
  int n = blockIdx.x, j = threadIdx.x;
  float acc = bias[j];
  #pragma unroll
  for (int k = 0; k < 4; ++k) acc += x[n * 4 + k] * W[k * HIDC + j];
  acc = bng[j] * (acc - bnm[j]) * rsqrtf(bnv[j] + 1e-5f) + bnb[j];
  acc = fmaxf(acc, 0.f);
  size_t o = (size_t)n * HIDC + j;
  h[o] = acc; hh[o] = (_Float16)acc;
}

// --------------------------- EdgeConv (WMMA) -------------------------------
// Per WG: 64 edges (4 M-tiles of 16), 8 waves = 8 N-tiles of 16 (N=128).
// Fused: async gather cat(h[src],h[dst]) -> GEMM1(K=256) -> BN+ReLU ->
//        GEMM2(K=128) -> BN+ReLU -> atomicAdd at src.
// Double-buffered sA: tile mt+1 gathers overlap tile mt compute.
__global__ __launch_bounds__(256)
void k_edgeconv(const _Float16* __restrict__ hh,
                const int* __restrict__ src, const int* __restrict__ dst,
                const _Float16* __restrict__ W1T,   // [128][256]
                const _Float16* __restrict__ W2T,   // [128][128]
                const float* __restrict__ lb1,
                const float* __restrict__ b1b, const float* __restrict__ b1g,
                const float* __restrict__ b1m, const float* __restrict__ b1v,
                const float* __restrict__ lb2,
                const float* __restrict__ b2b, const float* __restrict__ b2g,
                const float* __restrict__ b2m, const float* __restrict__ b2v,
                float* __restrict__ agg) {
  const int ST_A = 264;   // 256 + 8 halfs, keeps 16B alignment
  const int ST_M = 136;   // 128 + 8 halfs
  __shared__ _Float16 sA[2][16 * 264];
  __shared__ _Float16 sM[16 * 136];

  int tid = threadIdx.x;
  int wave = tid >> 5, lane = tid & 31;
  int laneN = lane & 15, hi = lane >> 4;
  int colg = wave * 16 + laneN;
  int e0 = blockIdx.x * 64;

  // B fragments in registers: K-halfs 8h..8h+7 and 16+8h..16+8h+7 per lane from WT[col].
  v16h B1[8], B2[4];
  {
    const _Float16* p1 = W1T + (size_t)colg * 256 + hi * 8;
    #pragma unroll
    for (int s = 0; s < 8; ++s) B1[s] = ldfrag(p1 + s * 32, p1 + s * 32 + 16);
    const _Float16* p2 = W2T + (size_t)colg * 128 + hi * 8;
    #pragma unroll
    for (int s = 0; s < 4; ++s) B2[s] = ldfrag(p2 + s * 32, p2 + s * 32 + 16);
  }
  float g1 = b1g[colg], c1b = b1b[colg], m1 = b1m[colg];
  float r1 = rsqrtf(b1v[colg] + 1e-5f), l1 = lb1[colg];
  float g2 = b2g[colg], c2b = b2b[colg], m2 = b2m[colg];
  float r2 = rsqrtf(b2v[colg] + 1e-5f), l2 = lb2[colg];

  // issue the 2 async gather chunks this thread owns for tile mt -> buf
  auto issue = [&](int mt, _Float16* buf) {
    #pragma unroll
    for (int i = 0; i < 2; ++i) {
      int c = tid + 256 * i;
      int row = c >> 5;
      int off = (c & 31) << 3;
      int e = e0 + mt * 16 + row;
      const _Float16* sp = (off < 128)
          ? (hh + (size_t)src[e] * HIDC + off)
          : (hh + (size_t)dst[e] * HIDC + (off - 128));
      async_g2l_b128(&buf[row * ST_A + off], sp);
    }
  };

  issue(0, sA[0]);
  for (int mt = 0; mt < 4; ++mt) {
    _Float16* cur = sA[mt & 1];
    if (mt < 3) {
      issue(mt + 1, sA[(mt + 1) & 1]);
      asm volatile("s_wait_asynccnt 0x2" ::: "memory");   // tile mt landed
    } else {
      asm volatile("s_wait_asynccnt 0x0" ::: "memory");
    }
    __syncthreads();

    // GEMM1: preload all 8 A fragments, then two interleaved WMMA chains.
    v16h Afr[8];
    #pragma unroll
    for (int s = 0; s < 8; ++s) {
      const _Float16* pa = &cur[laneN * ST_A + s * 32 + hi * 8];
      Afr[s] = ldfrag(pa, pa + 16);
    }
    v8f acc = {}, accB = {};
    #pragma unroll
    for (int s = 0; s < 8; s += 2) {
      acc  = __builtin_amdgcn_wmma_f32_16x16x32_f16(false, Afr[s],     false, B1[s],
                                                    (short)0, acc,  false, false);
      accB = __builtin_amdgcn_wmma_f32_16x16x32_f16(false, Afr[s + 1], false, B1[s + 1],
                                                    (short)0, accB, false, false);
    }
    #pragma unroll
    for (int r = 0; r < 8; ++r) acc[r] += accB[r];

    // BN1 + ReLU -> sM (f16, A-operand row-major)
    #pragma unroll
    for (int r = 0; r < 8; ++r) {
      float t = acc[r] + l1;
      t = fmaxf(g1 * (t - m1) * r1 + c1b, 0.f);
      sM[(r + hi * 8) * ST_M + colg] = (_Float16)t;
    }
    __syncthreads();

    v16h Mfr[4];
    #pragma unroll
    for (int s = 0; s < 4; ++s) {
      const _Float16* pa = &sM[laneN * ST_M + s * 32 + hi * 8];
      Mfr[s] = ldfrag(pa, pa + 16);
    }
    v8f acc2 = {};
    #pragma unroll
    for (int s = 0; s < 4; ++s)
      acc2 = __builtin_amdgcn_wmma_f32_16x16x32_f16(false, Mfr[s], false, B2[s],
                                                    (short)0, acc2, false, false);
    // BN2 + ReLU + scatter-add at src
    #pragma unroll
    for (int r = 0; r < 8; ++r) {
      float t = acc2[r] + l2;
      t = fmaxf(g2 * (t - m2) * r2 + c2b, 0.f);
      int e = e0 + mt * 16 + (r + hi * 8);
      atomicAdd(agg + (size_t)src[e] * HIDC + colg, t);
    }
  }
}

// ------------------------- add + LayerNorm (in place) ----------------------
__global__ __launch_bounds__(32)
void k_add_ln(const float* __restrict__ add, float* __restrict__ h,
              _Float16* __restrict__ hh,
              const float* __restrict__ g, const float* __restrict__ b) {
  int n = blockIdx.x, lane = threadIdx.x;
  float v[4], s = 0.f, s2 = 0.f;
  #pragma unroll
  for (int i = 0; i < 4; ++i) {
    size_t o = (size_t)n * HIDC + i * 32 + lane;
    float t = add[o] + h[o];
    v[i] = t; s += t; s2 += t * t;
  }
  #pragma unroll
  for (int m = 1; m < 32; m <<= 1) { s += __shfl_xor(s, m, 32); s2 += __shfl_xor(s2, m, 32); }
  float mu = s * (1.f / HIDC);
  float rs = rsqrtf(s2 * (1.f / HIDC) - mu * mu + 1e-5f);
  #pragma unroll
  for (int i = 0; i < 4; ++i) {
    int f = i * 32 + lane;
    size_t o = (size_t)n * HIDC + f;
    float t = g[f] * (v[i] - mu) * rs + b[f];
    h[o] = t; hh[o] = (_Float16)t;
  }
}

// ----------------------- GAT projection (WMMA) -----------------------------
__global__ __launch_bounds__(256)
void k_gat(const _Float16* __restrict__ hh, const _Float16* __restrict__ WT,
           const float* __restrict__ aS, const float* __restrict__ aD,
           float* __restrict__ xw, float* __restrict__ a_s,
           float* __restrict__ a_d, int nNodes) {
  int tid = threadIdx.x;
  int wave = tid >> 5, lane = tid & 31;
  int laneN = lane & 15, hi = lane >> 4;
  int col = wave * 16 + laneN;
  v16h B[4];
  const _Float16* bp = WT + (size_t)col * 128 + hi * 8;
  #pragma unroll
  for (int s = 0; s < 4; ++s) B[s] = ldfrag(bp + s * 32, bp + s * 32 + 16);
  float asw = aS[wave * 16 + laneN];
  float adw = aD[wave * 16 + laneN];

  int t0 = blockIdx.x * 4;
  for (int mt = 0; mt < 4; ++mt) {
    int nb = (t0 + mt) * 16;
    if (nb >= nNodes) break;
    const _Float16* pa0 = hh + (size_t)(nb + laneN) * HIDC + hi * 8;
    v16h Afr[4];
    #pragma unroll
    for (int s = 0; s < 4; ++s) Afr[s] = ldfrag(pa0 + s * 32, pa0 + s * 32 + 16);
    v8f acc = {};
    #pragma unroll
    for (int s = 0; s < 4; ++s)
      acc = __builtin_amdgcn_wmma_f32_16x16x32_f16(false, Afr[s], false, B[s],
                                                   (short)0, acc, false, false);
    #pragma unroll
    for (int r = 0; r < 8; ++r) {
      int row = nb + r + hi * 8;
      xw[(size_t)row * HIDC + col] = acc[r];
      float ps = acc[r] * asw, pd = acc[r] * adw;
      #pragma unroll
      for (int m = 1; m < 16; m <<= 1) { ps += __shfl_xor(ps, m, 16); pd += __shfl_xor(pd, m, 16); }
      if (laneN == 0) { a_s[(size_t)row * HEADSC + wave] = ps;
                        a_d[(size_t)row * HEADSC + wave] = pd; }
    }
  }
}

// ------------------------ edge softmax (3 passes) --------------------------
__global__ void k_att_max(const int* __restrict__ src, const int* __restrict__ dst,
                          const float* __restrict__ a_s, const float* __restrict__ a_d,
                          unsigned* __restrict__ mkey) {
  size_t i = (size_t)blockIdx.x * 256 + threadIdx.x;
  if (i >= (size_t)N_EDGESC * HEADSC) return;
  int e = (int)(i >> 3), hd = (int)(i & 7);
  float v = a_s[(size_t)src[e] * HEADSC + hd] + a_d[(size_t)dst[e] * HEADSC + hd];
  v = v > 0.f ? v : 0.2f * v;
  atomicMax(mkey + (size_t)dst[e] * HEADSC + hd, f2ord(v));
}

__global__ void k_att_exp(const int* __restrict__ src, const int* __restrict__ dst,
                          const float* __restrict__ a_s, const float* __restrict__ a_d,
                          const unsigned* __restrict__ mkey,
                          float* __restrict__ ssum, float* __restrict__ ex) {
  size_t i = (size_t)blockIdx.x * 256 + threadIdx.x;
  if (i >= (size_t)N_EDGESC * HEADSC) return;
  int e = (int)(i >> 3), hd = (int)(i & 7);
  int d = dst[e];
  float v = a_s[(size_t)src[e] * HEADSC + hd] + a_d[(size_t)d * HEADSC + hd];
  v = v > 0.f ? v : 0.2f * v;
  float m = ord2f(mkey[(size_t)d * HEADSC + hd]);
  float ev = expf(v - m);
  ex[i] = ev;
  atomicAdd(ssum + (size_t)d * HEADSC + hd, ev);
}

__global__ __launch_bounds__(128)
void k_att_scatter(const int* __restrict__ src, const int* __restrict__ dst,
                   const float* __restrict__ ex, const float* __restrict__ ssum,
                   const float* __restrict__ xw, float* __restrict__ att) {
  int e = blockIdx.x, f = threadIdx.x, hd = f >> 4;
  int s = src[e], d = dst[e];
  float alpha = ex[(size_t)e * HEADSC + hd] /
                (ssum[(size_t)d * HEADSC + hd] + 1e-16f);
  atomicAdd(att + (size_t)d * HIDC + f, xw[(size_t)s * HIDC + f] * alpha);
}

// ---------------- (att + bias + idn) -> LN -> (+idn) -> LN -----------------
__global__ __launch_bounds__(32)
void k_att_ln(const float* __restrict__ att, const float* __restrict__ gbias,
              float* __restrict__ h, _Float16* __restrict__ hh,
              const float* __restrict__ lab, const float* __restrict__ lag,
              const float* __restrict__ lob, const float* __restrict__ logg) {
  int n = blockIdx.x, lane = threadIdx.x;
  float idn[4], t[4], s = 0.f, s2 = 0.f;
  #pragma unroll
  for (int i = 0; i < 4; ++i) {
    int f = i * 32 + lane;
    size_t o = (size_t)n * HIDC + f;
    idn[i] = h[o];
    t[i] = att[o] + gbias[f] + idn[i];
    s += t[i]; s2 += t[i] * t[i];
  }
  #pragma unroll
  for (int m = 1; m < 32; m <<= 1) { s += __shfl_xor(s, m, 32); s2 += __shfl_xor(s2, m, 32); }
  float mu = s * (1.f / HIDC);
  float rs = rsqrtf(s2 * (1.f / HIDC) - mu * mu + 1e-5f);
  float u[4]; s = 0.f; s2 = 0.f;
  #pragma unroll
  for (int i = 0; i < 4; ++i) {
    int f = i * 32 + lane;
    float h1 = lag[f] * (t[i] - mu) * rs + lab[f];
    u[i] = h1 + idn[i];
    s += u[i]; s2 += u[i] * u[i];
  }
  #pragma unroll
  for (int m = 1; m < 32; m <<= 1) { s += __shfl_xor(s, m, 32); s2 += __shfl_xor(s2, m, 32); }
  float mu2 = s * (1.f / HIDC);
  float rs2 = rsqrtf(s2 * (1.f / HIDC) - mu2 * mu2 + 1e-5f);
  #pragma unroll
  for (int i = 0; i < 4; ++i) {
    int f = i * 32 + lane;
    size_t o = (size_t)n * HIDC + f;
    float v = logg[f] * (u[i] - mu2) * rs2 + lob[f];
    h[o] = v; hh[o] = (_Float16)v;
  }
}

// -------------------------- pooling accumulate -----------------------------
__global__ void k_pool(const float* __restrict__ h, const int* __restrict__ batch,
                       float* __restrict__ gsum, unsigned* __restrict__ gmax,
                       float* __restrict__ gcnt, int nNodes) {
  size_t i = (size_t)blockIdx.x * 256 + threadIdx.x;
  if (i >= (size_t)nNodes * HIDC) return;
  int n = (int)(i >> 7), f = (int)(i & 127);
  int g = batch[n];
  float v = h[i];
  atomicAdd(gsum + (size_t)g * HIDC + f, v);
  atomicMax(gmax + (size_t)g * HIDC + f, f2ord(v));
  if (f == 0) atomicAdd(gcnt + g, 1.0f);
}

// ---------------------------- MLP heads ------------------------------------
__global__ __launch_bounds__(128)
void k_head(const float* __restrict__ gsum, const unsigned* __restrict__ gmax,
            const float* __restrict__ gcnt,
            const float* __restrict__ Wp, const float* __restrict__ bp,
            const float* __restrict__ Wc1, const float* __restrict__ bc1,
            const float* __restrict__ Wc2, const float* __restrict__ bc2,
            const float* __restrict__ We1, const float* __restrict__ be1,
            const float* __restrict__ We2, const float* __restrict__ be2,
            float* __restrict__ out) {
  __shared__ float sin[256];
  __shared__ float sxg[128];
  __shared__ float sc1[64];
  __shared__ float se1[32];
  int g = blockIdx.x, t = threadIdx.x;
  float cnt = fmaxf(gcnt[g], 1.0f);
  sin[t] = gsum[(size_t)g * HIDC + t] / cnt;
  unsigned mk = gmax[(size_t)g * HIDC + t];
  sin[128 + t] = (mk == ORDNEG) ? 0.f : ord2f(mk);
  __syncthreads();
  float acc = bp[t];
  for (int k = 0; k < 256; ++k) acc += sin[k] * Wp[k * HIDC + t];
  sxg[t] = fmaxf(acc, 0.f);
  __syncthreads();
  if (t < 64) {
    float a = bc1[t];
    for (int k = 0; k < 128; ++k) a += sxg[k] * Wc1[k * 64 + t];
    sc1[t] = fmaxf(a, 0.f);
  } else if (t < 96) {
    int j = t - 64;
    float a = be1[j];
    for (int k = 0; k < 128; ++k) a += sxg[k] * We1[k * 32 + j];
    se1[j] = fmaxf(a, 0.f);
  }
  __syncthreads();
  if (t < N_CLASSESC) {
    float a = bc2[t];
    for (int k = 0; k < 64; ++k) a += sc1[k] * Wc2[k * N_CLASSESC + t];
    out[g * N_CLASSESC + t] = a;
  } else if (t == 8) {
    float a = be2[0];
    for (int k = 0; k < 32; ++k) a += se1[k] * We2[k];
    out[N_GRAPHSC * N_CLASSESC + g] = (a > 20.f) ? a : log1pf(expf(a));
  }
}

// ---------------------------------------------------------------------------
// Host: pytree leaf order assumed = top-level dict insertion order with
// params flattened alphabetically (jax tree_flatten):
//  0:x | per block (base=1+22*b): 0 ec1.W 1 ec1.b 2 ec2.W 3 ec2.b
//  4-7 ecbn1{b,g,m,v} 8-11 ecbn2{b,g,m,v} 12 gat.W 13 gat.a_dst 14 gat.a_src
//  15 gat.bias 16-17 ln_att{b,g} 18-19 ln_ec{b,g} 20-21 ln_out{b,g}
//  67 cls1.W 68 cls1.b 69 cls2.W 70 cls2.b 71 en1.W 72 en1.b 73 en2.W
//  74 en2.b 75 enc.W 76 enc.b 77-80 encbn{b,g,m,v} 81 pool.W 82 pool.b
//  83 edge_index 84 batch
// ---------------------------------------------------------------------------
extern "C" void kernel_launch(void* const* d_in, const int* in_sizes, int n_in,
                              void* d_out, int out_size, void* d_ws, size_t ws_size,
                              hipStream_t stream) {
  (void)in_sizes; (void)out_size; (void)ws_size;
  if (n_in < 85) return;
  const float* x = (const float*)d_in[0];
  const int* edge = (const int*)d_in[83];
  const int* srcI = edge;
  const int* dstI = edge + N_EDGESC;
  const int* batch = (const int*)d_in[84];
  auto F = [&](int i) { return (const float*)d_in[i]; };

  char* base = (char*)d_ws;
  size_t off = 0;
  auto alloc = [&](size_t bytes) -> void* {
    off = (off + 255) & ~(size_t)255;
    void* p = base + off;
    off += bytes;
    return p;
  };
  const size_t Np = 50048;  // N rounded up to 64
  float*    h    = (float*)alloc(Np * HIDC * 4);
  _Float16* hh   = (_Float16*)alloc(Np * HIDC * 2);
  float*    agg  = (float*)alloc(Np * HIDC * 4);   // also reused as att buffer
  float*    xw   = (float*)alloc(Np * HIDC * 4);
  float*    a_s  = (float*)alloc(Np * HEADSC * 4);
  float*    a_d  = (float*)alloc(Np * HEADSC * 4);
  unsigned* mkey = (unsigned*)alloc(Np * HEADSC * 4);
  float*    ssum = (float*)alloc(Np * HEADSC * 4);
  float*    ex   = (float*)alloc((size_t)N_EDGESC * HEADSC * 4);
  _Float16* W1T  = (_Float16*)alloc(256 * 128 * 2);
  _Float16* W2T  = (_Float16*)alloc(128 * 128 * 2);
  _Float16* WgT  = (_Float16*)alloc(128 * 128 * 2);
  float*    gsum = (float*)alloc(N_GRAPHSC * HIDC * 4);
  unsigned* gmax = (unsigned*)alloc(N_GRAPHSC * HIDC * 4);
  float*    gcnt = (float*)alloc(N_GRAPHSC * 4);

  const size_t NH = (size_t)N_NODESC * HIDC;       // 6.4M
  const size_t N8 = (size_t)N_NODESC * HEADSC;     // 400K
  const size_t E8 = (size_t)N_EDGESC * HEADSC;     // 5.12M
  auto blks = [](size_t n) { return (unsigned)((n + 255) / 256); };

  k_encoder<<<N_NODESC, 128, 0, stream>>>(x, F(75), F(76),
                                          F(77), F(78), F(79), F(80), h, hh);

  for (int b = 0; b < 3; ++b) {
    int pb = 1 + 22 * b;
    k_cvtT<<<blks(256 * 128), 256, 0, stream>>>(F(pb + 0), W1T, 256, 128);
    k_cvtT<<<blks(128 * 128), 256, 0, stream>>>(F(pb + 2), W2T, 128, 128);
    k_cvtT<<<blks(128 * 128), 256, 0, stream>>>(F(pb + 12), WgT, 128, 128);

    // --- EdgeConv ---
    k_fill<<<blks(NH), 256, 0, stream>>>((unsigned*)agg, 0u, NH);
    k_edgeconv<<<N_EDGESC / 64, 256, 0, stream>>>(
        hh, srcI, dstI, W1T, W2T,
        F(pb + 1), F(pb + 4), F(pb + 5), F(pb + 6), F(pb + 7),
        F(pb + 3), F(pb + 8), F(pb + 9), F(pb + 10), F(pb + 11), agg);
    k_add_ln<<<N_NODESC, 32, 0, stream>>>(agg, h, hh, F(pb + 19), F(pb + 18));

    // --- GAT ---
    k_gat<<<(3125 + 3) / 4, 256, 0, stream>>>(hh, WgT, F(pb + 14), F(pb + 13),
                                              xw, a_s, a_d, N_NODESC);
    k_fill<<<blks(N8), 256, 0, stream>>>(mkey, ORDNEG, N8);
    k_fill<<<blks(N8), 256, 0, stream>>>((unsigned*)ssum, 0u, N8);
    k_fill<<<blks(NH), 256, 0, stream>>>((unsigned*)agg, 0u, NH);  // att buf
    k_att_max<<<blks(E8), 256, 0, stream>>>(srcI, dstI, a_s, a_d, mkey);
    k_att_exp<<<blks(E8), 256, 0, stream>>>(srcI, dstI, a_s, a_d, mkey, ssum, ex);
    k_att_scatter<<<N_EDGESC, 128, 0, stream>>>(srcI, dstI, ex, ssum, xw, agg);
    k_att_ln<<<N_NODESC, 32, 0, stream>>>(agg, F(pb + 15), h, hh,
                                          F(pb + 16), F(pb + 17),
                                          F(pb + 20), F(pb + 21));
  }

  // --- pooling + heads ---
  k_fill<<<blks(N_GRAPHSC * HIDC), 256, 0, stream>>>((unsigned*)gsum, 0u, N_GRAPHSC * HIDC);
  k_fill<<<blks(N_GRAPHSC * HIDC), 256, 0, stream>>>(gmax, ORDNEG, N_GRAPHSC * HIDC);
  k_fill<<<1, 256, 0, stream>>>((unsigned*)gcnt, 0u, N_GRAPHSC);
  k_pool<<<blks(NH), 256, 0, stream>>>(h, batch, gsum, gmax, gcnt, N_NODESC);
  k_head<<<N_GRAPHSC, 128, 0, stream>>>(gsum, gmax, gcnt,
                                        F(81), F(82), F(67), F(68), F(69), F(70),
                                        F(71), F(72), F(73), F(74), (float*)d_out);
}
